// PPRDiffusedGCN_2164663517537
// MI455X (gfx1250) — matrix operand order repaired and checked by reading, not compiled, not run on hardware
//
#include <hip/hip_runtime.h>
#include <hip/hip_bf16.h>
#include <math.h>

typedef __attribute__((ext_vector_type(16))) _Float16 v16h;
typedef __attribute__((ext_vector_type(8)))  float    v8f;

#define NN     4096
#define DIN    128
#define DOUTF  512
#define NEDGE  65536
#define ALPHA_ 0.15f
#define KPOW   10
#define THRESH_ 1e-4f
#define LN_EPS_ 1e-5f

enum { MODE_PPR = 0, MODE_HHT = 1, MODE_MSG = 2 };

// Async global->LDS 16B copy (CDNA5 GLOBAL_LOAD_ASYNC_TO_LDS_B128, ASYNCcnt).
#define ASYNC_LDS_B128(ldsoff, gptr)                                          \
    asm volatile("global_load_async_to_lds_b128 %0, %1, off"                  \
                 :: "v"(ldsoff), "v"(gptr) : "memory")

// ---------------------------------------------------------------------------
// Edge scatter: adj[src][dst] += 1
// ---------------------------------------------------------------------------
__global__ void scatter_edges(const int* __restrict__ ei, float* __restrict__ adjcnt) {
    const int e = blockIdx.x * blockDim.x + threadIdx.x;
    if (e < NEDGE) {
        const int src = ei[e];
        const int dst = ei[NEDGE + e];
        atomicAdd(adjcnt + (size_t)src * NN + dst, 1.0f);
    }
}

// ---------------------------------------------------------------------------
// Row-normalize (with self loop), emit adj_norm (f16, row-major) and its
// transpose (f16) which doubles as the initial matT = (A^1)^T.
// ---------------------------------------------------------------------------
__global__ __launch_bounds__(256) void normalize_adj(const float* __restrict__ adjcnt,
                                                     _Float16* __restrict__ adjh,
                                                     _Float16* __restrict__ adjhT) {
    const int r = blockIdx.x;
    const int tid = threadIdx.x;
    __shared__ float red[256];
    float vals[16];
    float s = 0.f;
    const size_t base = (size_t)r * NN;
    #pragma unroll
    for (int j = 0; j < 16; ++j) {
        const int c = tid + j * 256;
        float v = adjcnt[base + c];
        if (c == r) v += 1.0f;           // self loop
        vals[j] = v;
        s += v;
    }
    red[tid] = s;
    __syncthreads();
    for (int off = 128; off > 0; off >>= 1) {
        if (tid < off) red[tid] += red[tid + off];
        __syncthreads();
    }
    const float inv = 1.0f / red[0];     // deg >= 1
    #pragma unroll
    for (int j = 0; j < 16; ++j) {
        const int c = tid + j * 256;
        const _Float16 h = (_Float16)(vals[j] * inv);
        adjh[base + c] = h;
        adjhT[(size_t)c * NN + r] = h;
    }
}

// pprT init: pprT = c0 * (float)adjT
__global__ void ppr_init(const _Float16* __restrict__ t, float* __restrict__ p, float c0) {
    const size_t i = ((size_t)blockIdx.x * blockDim.x + threadIdx.x) * 8;
    #pragma unroll
    for (int j = 0; j < 8; ++j) p[i + j] = c0 * (float)t[i + j];
}

// ---------------------------------------------------------------------------
// Tiled WMMA GEMM:  C[M][N] = A[M][K] @ B[K][N],  B given as BT[N][K].
// Block: 256 thr = 8 waves (2x4), tile 128x128, BK=32.
// Wave: 64x32 = 4x2 tiles of v_wmma_f32_16x16x32_f16.
// Tiles land in LDS *directly* via double-buffered GLOBAL_LOAD_ASYNC_TO_LDS,
// pre-permuted into CDNA5 WMMA fragment order (each lane's v16h fragment is
// one contiguous 32B run -> 2x ds_load_b128). Last iteration peeled so the
// steady-state loop body is branch-free.
// Epilogue modes:
//   MODE_PPR: CTh[N][M] = f16(C);  pprAcc[N][M] += coef*C
//   MODE_HHT: CTh[N][M] = f16(dinv[m] * C)
//   MODE_MSG: Crow[M][N] = dinv[m] * C
// ---------------------------------------------------------------------------
template<int MODE>
__global__ __launch_bounds__(256) void gemm_wmma_f16(
    const _Float16* __restrict__ A,
    const _Float16* __restrict__ BT,
    int M, int N, int K,
    _Float16* __restrict__ CTh,
    float* __restrict__ pprAcc,
    float coef,
    const float* __restrict__ dinv,
    float* __restrict__ Crow)
{
    __shared__ __align__(32) _Float16 sA[2][4096];  // 8KB per buffer
    __shared__ __align__(32) _Float16 sB[2][4096];

    const int tid  = threadIdx.x;
    const int lane = tid & 31;
    const int wave = tid >> 5;
    const int wmi  = wave >> 2;    // 0..1  (M direction, 64 rows each)
    const int wni  = wave & 3;     // 0..3  (N direction, 32 cols each)
    const int mBase = blockIdx.y * 128;
    const int nBase = blockIdx.x * 128;

    // ---- fixed per-thread staging assignment: 4 x 16B chunks per K-slab
    const int row = tid >> 2;            // 0..63 (and row+64)
    const int c   = tid & 3;             // 16B chunk along K
    // fragment-order LDS byte offsets (within one buffer)
    const int dA = ((row & 15) + ((c & 1) << 4)) * 32 + ((c >> 1) << 4);
    const int dB = ((row & 15) + ((c & 1) << 4)) * 32 + ((c >> 1) << 4);
    const unsigned sAbase = (unsigned)(uintptr_t)(&sA[0][0]);
    const unsigned sBbase = (unsigned)(uintptr_t)(&sB[0][0]);
    const unsigned ldsA0 = sAbase + (unsigned)((row >> 4) * 1024 + dA);
    const unsigned ldsA1 = ldsA0 + 4u * 1024u;   // rows 64..127 -> subtiles 4..7
    const unsigned ldsB0 = sBbase + (unsigned)((row >> 4) * 1024 + dB);
    const unsigned ldsB1 = ldsB0 + 4u * 1024u;
    const _Float16* gA0 = A  + (size_t)(mBase + row) * K + 8 * c;
    const _Float16* gA1 = gA0 + (size_t)64 * K;
    const _Float16* gB0 = BT + (size_t)(nBase + row) * K + 8 * c;
    const _Float16* gB1 = gB0 + (size_t)64 * K;

    v8f acc[4][2];
    #pragma unroll
    for (int a = 0; a < 4; ++a)
        #pragma unroll
        for (int b = 0; b < 2; ++b)
            acc[a][b] = (v8f){0.f, 0.f, 0.f, 0.f, 0.f, 0.f, 0.f, 0.f};

    // ---- prologue: tile 0 -> buffer 0
    ASYNC_LDS_B128(ldsA0, gA0);
    ASYNC_LDS_B128(ldsA1, gA1);
    ASYNC_LDS_B128(ldsB0, gB0);
    ASYNC_LDS_B128(ldsB1, gB1);

    const int numIt = K >> 5;
    // ---- steady state: branch-free body, last iteration peeled
    for (int it = 0; it < numIt - 1; ++it) {
        const int cur = it & 1;
        // issue next K-slab into the other buffer
        gA0 += 32; gA1 += 32; gB0 += 32; gB1 += 32;
        const unsigned bo = (unsigned)((cur ^ 1) * 8192);
        ASYNC_LDS_B128(ldsA0 + bo, gA0);
        ASYNC_LDS_B128(ldsA1 + bo, gA1);
        ASYNC_LDS_B128(ldsB0 + bo, gB0);
        ASYNC_LDS_B128(ldsB1 + bo, gB1);
        // retire previous tile's 4 ops (async loads complete in order)
        asm volatile("s_wait_asynccnt 4" ::: "memory");
        __syncthreads();   // buf[cur] visible to all waves

        v16h af[4], bf[2];
        #pragma unroll
        for (int a = 0; a < 4; ++a)
            af[a] = *(const v16h*)(sA[cur] + (wmi * 4 + a) * 512 + lane * 16);
        #pragma unroll
        for (int b = 0; b < 2; ++b)
            bf[b] = *(const v16h*)(sB[cur] + (wni * 2 + b) * 512 + lane * 16);

        #pragma unroll
        for (int a = 0; a < 4; ++a)
            #pragma unroll
            for (int b = 0; b < 2; ++b)
                acc[a][b] = __builtin_amdgcn_wmma_f32_16x16x32_f16(
                    false, af[a], false, bf[b], (short)0, acc[a][b], false, false);

        __syncthreads();   // all waves done reading buf[cur] before it is rewritten
    }

    // ---- final tile
    {
        const int cur = (numIt - 1) & 1;
        asm volatile("s_wait_asynccnt 0" ::: "memory");
        __syncthreads();

        v16h af[4], bf[2];
        #pragma unroll
        for (int a = 0; a < 4; ++a)
            af[a] = *(const v16h*)(sA[cur] + (wmi * 4 + a) * 512 + lane * 16);
        #pragma unroll
        for (int b = 0; b < 2; ++b)
            bf[b] = *(const v16h*)(sB[cur] + (wni * 2 + b) * 512 + lane * 16);

        #pragma unroll
        for (int a = 0; a < 4; ++a)
            #pragma unroll
            for (int b = 0; b < 2; ++b)
                acc[a][b] = __builtin_amdgcn_wmma_f32_16x16x32_f16(
                    false, af[a], false, bf[b], (short)0, acc[a][b], false, false);
    }

    // ---- epilogue: lane L holds N = L%16 (local), M = 8*(L/16) + r
    const int hl = lane >> 4;
    const int nl = lane & 15;
    #pragma unroll
    for (int a = 0; a < 4; ++a) {
        const int gm0 = mBase + wmi * 64 + a * 16 + hl * 8;
        #pragma unroll
        for (int b = 0; b < 2; ++b) {
            const int gn = nBase + wni * 32 + b * 16 + nl;
            if (MODE == MODE_PPR) {
                const size_t off = (size_t)gn * M + gm0;
                float*    p = pprAcc + off;
                _Float16* q = CTh + off;
                #pragma unroll
                for (int r = 0; r < 8; ++r) {
                    const float v = acc[a][b][r];
                    p[r] += coef * v;
                    q[r] = (_Float16)v;
                }
            } else if (MODE == MODE_HHT) {
                _Float16* q = CTh + (size_t)gn * M + gm0;
                #pragma unroll
                for (int r = 0; r < 8; ++r)
                    q[r] = (_Float16)(dinv[gm0 + r] * acc[a][b][r]);
            } else {
                #pragma unroll
                for (int r = 0; r < 8; ++r)
                    Crow[(size_t)(gm0 + r) * N + gn] = dinv[gm0 + r] * acc[a][b][r];
            }
        }
    }
}

// ---------------------------------------------------------------------------
// Finalize: pprT += alpha*I (diag), threshold, f16 copy, row sums -> dinv.
// ---------------------------------------------------------------------------
__global__ __launch_bounds__(256) void finalize_ppr(float* __restrict__ pprT,
                                                    _Float16* __restrict__ pprh,
                                                    float* __restrict__ dinv) {
    const int n = blockIdx.x;
    const int tid = threadIdx.x;
    __shared__ float red[256];
    const size_t base = (size_t)n * NN;
    float s = 0.f;
    #pragma unroll
    for (int j = 0; j < 16; ++j) {
        const int m = tid + j * 256;
        float v = pprT[base + m];
        if (m == n) v += ALPHA_;
        v = (v < THRESH_) ? 0.f : v;
        pprT[base + m] = v;
        pprh[base + m] = (_Float16)v;
        s += v;
    }
    red[tid] = s;
    __syncthreads();
    for (int off = 128; off > 0; off >>= 1) {
        if (tid < off) red[tid] += red[tid + off];
        __syncthreads();
    }
    if (tid == 0) {
        const float d = red[0];
        dinv[n] = (d > 0.f) ? rsqrtf(d) : 0.f;
    }
}

__global__ void f32_to_f16_kernel(const float* __restrict__ in, _Float16* __restrict__ out) {
    const size_t i = (size_t)blockIdx.x * blockDim.x + threadIdx.x;
    out[i] = (_Float16)in[i];
}

// W[K][N] -> WT[N][K] as f16
__global__ void transpose_f32_to_f16(const float* __restrict__ in, _Float16* __restrict__ out,
                                     int Kd, int Nd) {
    const int i = blockIdx.x * blockDim.x + threadIdx.x;  // over Nd*Kd
    const int n = i / Kd;
    const int k = i - n * Kd;
    out[i] = (_Float16)in[(size_t)k * Nd + n];
}

// ---------------------------------------------------------------------------
// out = ELU( LN(msg + b) * gamma + beta ), in-place over d_out rows.
// ---------------------------------------------------------------------------
__global__ __launch_bounds__(256) void ln_elu(float* __restrict__ out,
                                              const float* __restrict__ bias,
                                              const float* __restrict__ gamma,
                                              const float* __restrict__ beta) {
    const int m = blockIdx.x;
    const int tid = threadIdx.x;
    __shared__ float rs_[256], rq_[256];
    const size_t base = (size_t)m * DOUTF;
    const float v0 = out[base + tid] + bias[tid];
    const float v1 = out[base + tid + 256] + bias[tid + 256];
    rs_[tid] = v0 + v1;
    rq_[tid] = v0 * v0 + v1 * v1;
    __syncthreads();
    for (int off = 128; off > 0; off >>= 1) {
        if (tid < off) { rs_[tid] += rs_[tid + off]; rq_[tid] += rq_[tid + off]; }
        __syncthreads();
    }
    const float mu  = rs_[0] * (1.0f / DOUTF);
    const float var = rq_[0] * (1.0f / DOUTF) - mu * mu;
    const float inv = rsqrtf(var + LN_EPS_);
    float y0 = (v0 - mu) * inv * gamma[tid] + beta[tid];
    float y1 = (v1 - mu) * inv * gamma[tid + 256] + beta[tid + 256];
    y0 = (y0 > 0.f) ? y0 : expm1f(y0);
    y1 = (y1 > 0.f) ? y1 : expm1f(y1);
    out[base + tid] = y0;
    out[base + tid + 256] = y1;
}

// ---------------------------------------------------------------------------
extern "C" void kernel_launch(void* const* d_in, const int* in_sizes, int n_in,
                              void* d_out, int out_size, void* d_ws, size_t ws_size,
                              hipStream_t stream) {
    (void)in_sizes; (void)n_in; (void)out_size; (void)ws_size;
    const float* x     = (const float*)d_in[0];
    const int*   ei    = (const int*)d_in[1];
    const float* W     = (const float*)d_in[2];
    const float* b     = (const float*)d_in[3];
    const float* gamma = (const float*)d_in[4];
    const float* beta  = (const float*)d_in[5];

    char* w = (char*)d_ws;
    size_t off = 0;
    auto alloc = [&](size_t bytes) -> void* {
        void* p = w + off;
        off = (off + bytes + 255) & ~(size_t)255;
        return p;
    };
    float*    pprT   = (float*)alloc((size_t)NN * NN * 4);      // 64 MB, [n][m] = ppr[m][n]
    float*    adjcnt = (float*)alloc((size_t)NN * NN * 4);      // 64 MB
    _Float16* adjh   = (_Float16*)alloc((size_t)NN * NN * 2);   // 32 MB, reused as pprh
    _Float16* matT0  = (_Float16*)alloc((size_t)NN * NN * 2);   // 32 MB, starts as adj^T
    _Float16* matT1  = (_Float16*)alloc((size_t)NN * NN * 2);   // 32 MB
    _Float16* hhT    = (_Float16*)alloc((size_t)DOUTF * NN * 2);// 4 MB
    _Float16* xh     = (_Float16*)alloc((size_t)NN * DIN * 2);
    _Float16* WTh    = (_Float16*)alloc((size_t)DOUTF * DIN * 2);
    float*    dinv   = (float*)alloc((size_t)NN * 4);

    // 1) dense adjacency
    hipMemsetAsync(adjcnt, 0, (size_t)NN * NN * 4, stream);
    scatter_edges<<<NEDGE / 256, 256, 0, stream>>>(ei, adjcnt);
    normalize_adj<<<NN, 256, 0, stream>>>(adjcnt, adjh, matT0);

    // 2) PPR power series (fused accumulate into pprT)
    ppr_init<<<(NN * (size_t)NN) / (8 * 256), 256, 0, stream>>>(matT0, pprT, 1.0f - ALPHA_);
    _Float16* cur = matT0;
    _Float16* nxt = matT1;
    float co = 1.0f - ALPHA_;
    for (int i = 1; i < KPOW; ++i) {
        co *= ALPHA_;
        gemm_wmma_f16<MODE_PPR><<<dim3(NN / 128, NN / 128), 256, 0, stream>>>(
            adjh, cur, NN, NN, NN, nxt, pprT, co, nullptr, nullptr);
        _Float16* t = cur; cur = nxt; nxt = t;
    }

    // 3) threshold + alpha*I + degrees
    finalize_ppr<<<NN, 256, 0, stream>>>(pprT, adjh /*-> pprh*/, dinv);

    // 4) hhT[n][m] = f16(dinv[m] * (x@W)[m][n])
    f32_to_f16_kernel<<<(NN * DIN) / 256, 256, 0, stream>>>(x, xh);
    transpose_f32_to_f16<<<(DOUTF * DIN) / 256, 256, 0, stream>>>(W, WTh, DIN, DOUTF);
    gemm_wmma_f16<MODE_HHT><<<dim3(DOUTF / 128, NN / 128), 256, 0, stream>>>(
        xh, WTh, NN, DOUTF, DIN, hhT, nullptr, 0.f, dinv, nullptr);

    // 5) msg[m][n] = dinv[m] * (pprT_mat @ hh)[m][n]  -> d_out
    gemm_wmma_f16<MODE_MSG><<<dim3(DOUTF / 128, NN / 128), 256, 0, stream>>>(
        adjh /*pprh*/, hhT, NN, DOUTF, NN, nullptr, nullptr, 0.f, dinv, (float*)d_out);

    // 6) +b, LayerNorm, ELU (in place)
    ln_elu<<<NN, 256, 0, stream>>>((float*)d_out, b, gamma, beta);
}
